// PyUriTwc_24309514895623
// MI455X (gfx1250) — matrix operand order repaired and checked by reading, not compiled, not run on hardware
//
#include <hip/hip_runtime.h>
#include <math.h>

typedef __attribute__((ext_vector_type(16))) _Float16 v16h;
typedef __attribute__((ext_vector_type(8)))  _Float16 v8h;
typedef __attribute__((ext_vector_type(8)))  float    v8f;
typedef __attribute__((ext_vector_type(4)))  unsigned int u32x4;
typedef __attribute__((ext_vector_type(8)))  int          i32x8;
typedef __attribute__((ext_vector_type(4)))  int          i32x4;

#define N_NEUR 2048
#define BATCH  8
#define TSTEPS 4
#define NIN    32
#define NOUT   8
#define TILE_D 16
#define TILE_S 256
#define WG_STRIDE 257   // TDM pad: 1 DWORD after every 256 DWORDs -> stride 257

// LDS layout (bytes)
#define OFF_O32   0                         // 8*2048 f32   = 65536
#define OFF_O16   65536                     // 9*2048 f16   = 36864 (row 8 = zeros)
#define OFF_WG    (65536 + 36864)           // 16*257 f32   = 16448
#define OFF_CRED  (OFF_WG + 16448)          // 8*256 f32    = 8192
#define OFF_GRED  (OFF_CRED + 8192)         // 256 f32      = 1024
#define OFF_EH    (OFF_GRED + 1024)         // 128 f32      = 512
#define SMEM_BYTES (OFF_EH + 512)

__device__ __forceinline__ float softplusf(float x) {
    return (x > 0.f) ? (x + log1pf(expf(-x))) : log1pf(expf(x));
}

// --- Preprocess: W_pos = softplus(w); W_chem (f16) and W_gj (f32) ------------
__global__ void prep_kernel(const float* __restrict__ w,
                            const float* __restrict__ mex,
                            const float* __restrict__ min_,
                            const float* __restrict__ mgj,
                            _Float16* __restrict__ Wc16,
                            float* __restrict__ Wgj) {
    size_t i = (size_t)blockIdx.x * blockDim.x + threadIdx.x;
    float sp = softplusf(w[i]);
    Wc16[i]  = (_Float16)(sp * (mex[i] - min_[i]));
    Wgj[i]   = sp * mgj[i];
}

__global__ void zero_kernel(float* __restrict__ p) {
    p[(size_t)blockIdx.x * blockDim.x + threadIdx.x] = 0.f;
}

// --- One recurrent step ------------------------------------------------------
__global__ __launch_bounds__(256)
void step_kernel(const float* __restrict__ obs,      // (B, T, NIN)
                 const _Float16* __restrict__ Wc16,  // (N, N) f16 row-major
                 const float* __restrict__ Wgj,      // (N, N) f32 row-major
                 const float* __restrict__ thr,      // (N)
                 const float* __restrict__ dec,      // (N)
                 const float* __restrict__ Eprev,    // (B, N)
                 const float* __restrict__ Oprev,    // (B, N)
                 float* __restrict__ Enext,
                 float* __restrict__ Onext,
                 float* __restrict__ out,            // (B, T, NOUT)
                 int t) {
    extern __shared__ char smem[];
    float*    O_lds    = (float*)(smem + OFF_O32);
    _Float16* O16      = (_Float16*)(smem + OFF_O16);
    float*    Wg_lds   = (float*)(smem + OFF_WG);
    float*    chem_red = (float*)(smem + OFF_CRED);
    float*    gj_red   = (float*)(smem + OFF_GRED);
    float*    E_lds    = (float*)(smem + OFF_EH);

    const int tid    = threadIdx.x;
    const int d_base = blockIdx.x * TILE_D;

    // 1) Stage O_h (with input override) to LDS: f32 for the sign path,
    //    f16 copy (+1 zero pad row) for the WMMA A operand.
    for (int idx = tid; idx < BATCH * N_NEUR; idx += 256) {
        int b = idx >> 11, s = idx & (N_NEUR - 1);
        float v = (s < NIN) ? obs[b * (TSTEPS * NIN) + t * NIN + s] : Oprev[idx];
        O_lds[idx] = v;
        O16[idx]   = (_Float16)v;
    }
    for (int idx = tid; idx < N_NEUR; idx += 256)
        O16[BATCH * N_NEUR + idx] = (_Float16)0.f;   // zero row for pad lanes

    // 2) E_h for this block's 16 neurons
    if (tid < BATCH * TILE_D) {
        int b = tid >> 4, dl = tid & 15, d = d_base + dl;
        float v = (d < NIN) ? obs[b * (TSTEPS * NIN) + t * NIN + d]
                            : Eprev[b * N_NEUR + d];
        E_lds[tid] = v;
    }
    __syncthreads();

    // 3) I_chem = O_h @ W_chem.T via v_wmma_f32_16x16x32_f16.
    //    Wave w owns K-slice [w*256, w*256+256); partial C tiles reduced in LDS.
    const int lane = tid & 31;
    const int wave = tid >> 5;
    const int mcol = lane & 15;          // A row m / B col n
    const int hi   = lane >> 4;
    const _Float16* orow16 = O16 + (mcol < BATCH ? mcol : BATCH) * N_NEUR;
    const _Float16* wrow   = Wc16 + (size_t)(d_base + mcol) * N_NEUR;

    v8f c = {};
    for (int kb = wave * 256; kb < wave * 256 + 256; kb += 32) {
        // A fragment (16-bit A 16x32 layout): 2x ds_load_b128
        v8h a0 = *(const v8h*)(orow16 + kb + hi * 8);
        v8h a1 = *(const v8h*)(orow16 + kb + 16 + hi * 8);
        // B fragment: lane n holds K = hi*8..hi*8+7, 16+hi*8..16+hi*8+7: 2x global b128
        v8h b0 = *(const v8h*)(wrow + kb + hi * 8);
        v8h b1 = *(const v8h*)(wrow + kb + 16 + hi * 8);
        v16h a, bf;
        #pragma unroll
        for (int h = 0; h < 8; ++h) {
            a[h] = a0[h];  a[8 + h] = a1[h];
            bf[h] = b0[h]; bf[8 + h] = b1[h];
        }
        c = __builtin_amdgcn_wmma_f32_16x16x32_f16(
                false, a, false, bf, (short)0, c, false, false);
    }
    #pragma unroll
    for (int r = 0; r < 8; ++r) {        // C/D layout: vgpr r -> M=r (+8 hi)
        int mm = hi ? (r + 8) : r;
        chem_red[wave * 256 + mm * 16 + mcol] = c[r];
    }
    __syncthreads();

    float Ichem = 0.f;
    if (tid < BATCH * TILE_D) {
        int b = tid >> 4, dl = tid & 15;
        #pragma unroll
        for (int w = 0; w < 8; ++w) Ichem += chem_red[w * 256 + b * 16 + dl];
    }

    // 4) I_gj: sign-coupled reduction. W_gj tiles staged by the Tensor Data
    //    Mover (TDM) with hardware LDS padding (256 DW interval, 1 DW amount)
    //    reproducing the 257-float bank-skewed row stride.
    const int p   = tid & 127;
    const int par = tid >> 7;
    const int bb  = p >> 4;
    const int dl2 = p & 15;
    const float Eh = E_lds[bb * 16 + dl2];
    float acc = 0.f;

    for (int tile = 0; tile < N_NEUR / TILE_S; ++tile) {
        __syncthreads();                 // previous tile fully consumed
        if (tid < 32) {                  // wave 0 issues the DMA
            unsigned long long ga = (unsigned long long)(const void*)
                (Wgj + (size_t)d_base * N_NEUR + (size_t)tile * TILE_S);
            u32x4 g0;
            g0[0] = 1u;                                       // count=1
            g0[1] = (unsigned)OFF_WG;                         // lds_addr
            g0[2] = (unsigned)(ga & 0xFFFFFFFFu);             // global_addr lo
            g0[3] = (unsigned)((ga >> 32) & 0x1FFFFFFu)       // global_addr hi
                    | (2u << 30);                             // type = image
            i32x8 g1;
            g1[0] = (int)((2u << 16)     // data_size = 4B
                        | (1u << 20)     // pad_enable
                        | (7u << 22));   // pad_interval = 256 DW, amount = 1 DW
            g1[1] = (int)(((unsigned)N_NEUR & 0xFFFFu) << 16);        // dim0 lo
            g1[2] = (int)((((unsigned)N_NEUR >> 16) & 0xFFFFu)
                        | (((unsigned)N_NEUR & 0xFFFFu) << 16));      // dim0 hi | dim1 lo
            g1[3] = (int)((((unsigned)N_NEUR >> 16) & 0xFFFFu)
                        | ((unsigned)TILE_S << 16));                  // dim1 hi | tile_dim0
            g1[4] = TILE_D;                                           // tile_dim1
            g1[5] = N_NEUR;                                           // dim0_stride lo
            g1[6] = (int)(((unsigned)N_NEUR & 0xFFFFu) << 16);        // dim1_stride lo
            g1[7] = 0;
            i32x4 z4 = {0, 0, 0, 0};
            i32x8 z8 = {0, 0, 0, 0, 0, 0, 0, 0};
            __builtin_amdgcn_tensor_load_to_lds(g0, g1, z4, z4, z8, 0);
            __builtin_amdgcn_s_wait_tensorcnt(0);
        }
        __syncthreads();                 // tile visible to all waves

        const float* ob = O_lds + bb * N_NEUR + tile * TILE_S;
        const float* wg = Wg_lds + dl2 * WG_STRIDE;
        #pragma unroll 8
        for (int i = 0; i < TILE_S / 2; ++i) {
            int s = 2 * i + par;
            float O = ob[s];
            float W = wg[s];
            float diff = O - Eh;
            float wo = W * O;
            acc += (diff > 0.f) ? wo : ((diff < 0.f) ? -wo : 0.f);
        }
    }
    __syncthreads();
    gj_red[tid] = acc;
    __syncthreads();

    // 5) State update + outputs
    if (tid < BATCH * TILE_D) {
        float Igj = gj_red[tid] + gj_red[tid + 128];
        int b = tid >> 4, dl = tid & 15, d = d_base + dl;
        float Eh2 = E_lds[tid];
        float curr = Eh2 + Ichem + Igj;
        curr = fminf(10.f, fmaxf(-10.f, curr));
        float th = thr[d];
        bool firing = curr > th;
        float vf = curr - th;
        bool dm = (!firing) && (fabsf(Eh2 - curr) < 1e-5f);
        float En = firing ? vf : (dm ? (Eh2 - dec[d]) : curr);
        float On = firing ? vf : 0.f;
        Enext[b * N_NEUR + d] = En;      // input cols re-overridden next step
        Onext[b * N_NEUR + d] = On;
        if (d >= N_NEUR - NOUT)
            out[b * (TSTEPS * NOUT) + t * NOUT + (d - (N_NEUR - NOUT))] = En;
    }
}

extern "C" void kernel_launch(void* const* d_in, const int* in_sizes, int n_in,
                              void* d_out, int out_size, void* d_ws, size_t ws_size,
                              hipStream_t stream) {
    (void)in_sizes; (void)n_in; (void)out_size; (void)ws_size;
    const float* obs     = (const float*)d_in[0];
    const float* weights = (const float*)d_in[1];
    const float* thr     = (const float*)d_in[2];
    const float* dec     = (const float*)d_in[3];
    const float* mex     = (const float*)d_in[4];
    const float* min_    = (const float*)d_in[5];
    const float* mgj     = (const float*)d_in[6];
    float* out = (float*)d_out;

    char* ws = (char*)d_ws;
    _Float16* Wc16 = (_Float16*)ws;                                  // 8 MB
    float* Wgj  = (float*)(ws + (size_t)N_NEUR * N_NEUR * 2);        // 16 MB
    float* st   = (float*)(ws + (size_t)N_NEUR * N_NEUR * 6);        // state
    float* E0 = st;
    float* E1 = E0 + BATCH * N_NEUR;
    float* O0 = E1 + BATCH * N_NEUR;
    float* O1 = O0 + BATCH * N_NEUR;

    prep_kernel<<<(N_NEUR * N_NEUR) / 256, 256, 0, stream>>>(
        weights, mex, min_, mgj, Wc16, Wgj);
    zero_kernel<<<(4 * BATCH * N_NEUR) / 256, 256, 0, stream>>>(st);

    for (int t = 0; t < TSTEPS; ++t) {
        const float* Ep = (t & 1) ? E1 : E0;
        const float* Op = (t & 1) ? O1 : O0;
        float* En = (t & 1) ? E0 : E1;
        float* On = (t & 1) ? O0 : O1;
        step_kernel<<<N_NEUR / TILE_D, 256, SMEM_BYTES, stream>>>(
            obs, Wc16, Wgj, thr, dec, Ep, Op, En, On, out, t);
    }
}
